// ProgressiveContrastiveLearning_21517786153384
// MI455X (gfx1250) — matrix operand order repaired and checked by reading, not compiled, not run on hardware
//
#include <hip/hip_runtime.h>
#include <hip/hip_bf16.h>
#include <math.h>

#define BN 8192
#define DK 1024
#define INV_T (1.0f / 0.07f)
#define EPSV 1e-8f

typedef __attribute__((ext_vector_type(16))) _Float16 v16h;
typedef __attribute__((ext_vector_type(8)))  _Float16 v8h;
typedef __attribute__((ext_vector_type(8)))  float    v8f;

union V16 { v16h v; v8h h[2]; };

// ---------------------------------------------------------------------------
// Kernel 1: L2-normalize rows of visual / textual feats, emit f16 copies.
// One 256-thread block per row; blocks [0,8192) -> visual, [8192,16384) -> text.
// ---------------------------------------------------------------------------
__global__ __launch_bounds__(256)
void pcl_norm_f16_kernel(const float* __restrict__ vis,
                         const float* __restrict__ tex,
                         _Float16* __restrict__ vh,
                         _Float16* __restrict__ th) {
    __shared__ float red[256];
    const int row = blockIdx.x;
    const float* src;
    _Float16* dst;
    if (row < BN) {
        src = vis + (size_t)row * DK;
        dst = vh  + (size_t)row * DK;
    } else {
        src = tex + (size_t)(row - BN) * DK;
        dst = th  + (size_t)(row - BN) * DK;
    }
    const int tid = threadIdx.x;
    float x0 = src[tid];
    float x1 = src[tid + 256];
    float x2 = src[tid + 512];
    float x3 = src[tid + 768];
    red[tid] = x0 * x0 + x1 * x1 + x2 * x2 + x3 * x3;
    __syncthreads();
#pragma unroll
    for (int s = 128; s > 0; s >>= 1) {
        if (tid < s) red[tid] += red[tid + s];
        __syncthreads();
    }
    const float inv = 1.0f / fmaxf(sqrtf(red[0]), 1e-12f);
    dst[tid]       = (_Float16)(x0 * inv);
    dst[tid + 256] = (_Float16)(x1 * inv);
    dst[tid + 512] = (_Float16)(x2 * inv);
    dst[tid + 768] = (_Float16)(x3 * inv);
}

// ---------------------------------------------------------------------------
// Kernel 2: seed per-row accumulators with EPS.
// ---------------------------------------------------------------------------
__global__ void pcl_init_negsum_kernel(float* __restrict__ neg_sum) {
    neg_sum[blockIdx.x * 256 + threadIdx.x] = EPSV;
}

// ---------------------------------------------------------------------------
// Kernel 3: f16 WMMA GEMM (sim = vh * th^T), fused mask * exp(sim/T) and
// per-row sum.
//   Block = 8 waves (256 thr). Wave tile = 64(M) x 64(N) = 4x4 WMMA subtiles.
//   Block tile = 128(M) x 256(N)  -> L2 traffic: A 512MB + B 1GB total.
//   16 WMMAs per 16 b128 loads per k-step of 32.
// A/B fragments stream straight from row-major global in a contiguous k order (k = 16*half + s, same for A and B)
// (a B column == a th row, contiguous), so no LDS, no transpose.
// ---------------------------------------------------------------------------
__global__ __launch_bounds__(256)
void pcl_gemm_exp_rowsum_kernel(const _Float16* __restrict__ vh,
                                const _Float16* __restrict__ th,
                                const int* __restrict__ noisy,
                                float* __restrict__ neg_sum) {
    const int lane = threadIdx.x & 31;
    const int wave = threadIdx.x >> 5;
    const int half = lane >> 4;   // 0: lanes 0-15, 1: lanes 16-31
    const int l16  = lane & 15;
    const int wm   = wave >> 2;   // 0..1
    const int wn   = wave & 3;    // 0..3

    const int rowA = blockIdx.y * 128 + wm * 64;   // wave rows: +0 .. +63
    const int colB = blockIdx.x * 256 + wn * 64;   // wave cols: +0 .. +63

    v8f acc[4][4] = {};

    // A fragment: lane l<16 owns row (rowA + mi*16 + l), K {k..k+15};
    //             lane l+16 owns same row,               K {k+16..k+31}.
    const _Float16* aptr[4];
    // B fragment: lane owns th row (colB + ni*16 + l16), 16 contiguous f16
    // starting at k + half*16.
    const _Float16* bptr[4];
#pragma unroll
    for (int t = 0; t < 4; ++t) {
        aptr[t] = vh + (size_t)(rowA + t * 16 + l16) * DK + half * 16;
        bptr[t] = th + (size_t)(colB + t * 16 + l16) * DK + half * 16;
    }

    for (int k = 0; k < DK; k += 32) {
        V16 a[4];
        v16h b[4];
#pragma unroll
        for (int t = 0; t < 4; ++t) {
            a[t].h[0] = *(const v8h*)(aptr[t] + k);
            a[t].h[1] = *(const v8h*)(aptr[t] + k + 8);
            b[t]      = *(const v16h*)(bptr[t] + k);
        }
        // Pull the next cachelines toward the caches (global_prefetch_b8).
#pragma unroll
        for (int t = 0; t < 4; ++t) {
            __builtin_prefetch(aptr[t] + k + 128, 0, 1);
            __builtin_prefetch(bptr[t] + k + 128, 0, 1);
        }
#pragma unroll
        for (int mi = 0; mi < 4; ++mi)
#pragma unroll
            for (int ni = 0; ni < 4; ++ni)
                acc[mi][ni] = __builtin_amdgcn_wmma_f32_16x16x32_f16(
                    false, a[mi].v, false, b[ni], (short)0, acc[mi][ni],
                    false, false);
    }

    // Epilogue: C layout -> VGPR r holds row (base + r + 8*half), col = l16.
    bool nj[4];
#pragma unroll
    for (int t = 0; t < 4; ++t) nj[t] = noisy[colB + t * 16 + l16] != 0;

#pragma unroll
    for (int mi = 0; mi < 4; ++mi) {
        const int i0 = rowA + mi * 16 + half * 8;
#pragma unroll
        for (int r = 0; r < 8; ++r) {
            const int i = i0 + r;
            const bool ni = noisy[i] != 0;
            float s = 0.0f;
#pragma unroll
            for (int t = 0; t < 4; ++t)
                s += (ni || nj[t]) ? __expf(acc[mi][t][r] * INV_T) : 0.0f;
            // reduce the 16 lanes of this half-wave (xor stays within a half)
            s += __shfl_xor(s, 1, 32);
            s += __shfl_xor(s, 2, 32);
            s += __shfl_xor(s, 4, 32);
            s += __shfl_xor(s, 8, 32);
            if (l16 == 0) atomicAdd(&neg_sum[i], s);
        }
    }
}

// ---------------------------------------------------------------------------
// Kernel 4: loss = mean(log(neg_sum))
// ---------------------------------------------------------------------------
__global__ __launch_bounds__(256)
void pcl_final_loss_kernel(const float* __restrict__ neg_sum,
                           float* __restrict__ out) {
    __shared__ float red[256];
    const int tid = threadIdx.x;
    float s = 0.0f;
    for (int i = tid; i < BN; i += 256) s += __logf(neg_sum[i]);
    red[tid] = s;
    __syncthreads();
#pragma unroll
    for (int st = 128; st > 0; st >>= 1) {
        if (tid < st) red[tid] += red[tid + st];
        __syncthreads();
    }
    if (tid == 0) out[0] = red[0] / (float)BN;
}

extern "C" void kernel_launch(void* const* d_in, const int* in_sizes, int n_in,
                              void* d_out, int out_size, void* d_ws, size_t ws_size,
                              hipStream_t stream) {
    const float* vis = (const float*)d_in[0];
    const float* tex = (const float*)d_in[1];
    // d_in[2] transport_matrix   : unused by the loss (dead input)
    // d_in[3] confident_clean    : unused in push_only phase
    const int* noisy = (const int*)d_in[4];

    char* ws = (char*)d_ws;
    _Float16* vh      = (_Float16*)ws;                             // 16 MB
    _Float16* th      = (_Float16*)(ws + (size_t)BN * DK * 2);     // 16 MB
    float*    neg_sum = (float*)   (ws + (size_t)BN * DK * 4);     // 32 KB

    pcl_norm_f16_kernel<<<2 * BN, 256, 0, stream>>>(vis, tex, vh, th);
    pcl_init_negsum_kernel<<<BN / 256, 256, 0, stream>>>(neg_sum);

    dim3 grid(BN / 256, BN / 128);  // x: col blocks (256 wide), y: row blocks (128 tall)
    pcl_gemm_exp_rowsum_kernel<<<grid, 256, 0, stream>>>(vh, th, noisy, neg_sum);

    pcl_final_loss_kernel<<<1, 256, 0, stream>>>(neg_sum, (float*)d_out);
}
